// MultiAttention_34514357190902
// MI455X (gfx1250) — compile-verified, matrix operand order checked by reading, not run on hardware
//
#include <hip/hip_runtime.h>
#include <hip/hip_bf16.h>

// MI455X / gfx1250, wave32. bf16 WMMA for projections AND the MLP scorer
// (refactored as Msk@P^T + E@Q^T), f32 accumulation throughout.
// Bv fragments staged through LDS to keep peak VGPRs < 256.

typedef __attribute__((ext_vector_type(16))) __bf16 bf16x16;
typedef __attribute__((ext_vector_type(2)))  __bf16 bf16x2;
typedef __attribute__((ext_vector_type(8)))  float  f32x8;

#define N_BATCH 128
#define L_SEQ   1024
#define D_EMB   64
#define H_HEADS 4
#define HD      256
#define TL      256           // L rows per workgroup
#define RB      16            // rows per WMMA block
#define NBLK    (TL / RB)
#define ES_STRIDE 66          // even padded stride -> 8B-aligned float2 LDS loads

// f32 pair -> packed bf16 via native __bf16 conversion (RNE; hw cvt on gfx1250)
__device__ __forceinline__ unsigned int pack2_bf16(float a, float b) {
  bf16x2 p;
  p.x = (__bf16)a;
  p.y = (__bf16)b;
  return __builtin_bit_cast(unsigned int, p);
}

// 16-bit A/B fragment K index mapping for v_wmma_*_16x16x32 (ISA 7.12.2):
// VGPR r<4  : K = 2r,2r+1   (+8 for lanes 16..31)
// VGPR r>=4 : K = 16+2(r-4) (+8 for lanes 16..31)
__device__ __forceinline__ int kmap(int r, int half) {
  return ((r < 4) ? (2 * r) : (16 + 2 * (r - 4))) + half * 8;
}

__device__ __forceinline__ float sigmoidf_fast(float x) {
  return 1.0f / (1.0f + __expf(-x));
}

// ---------------------------------------------------------------------------
// prep: vproj[n][c] = sum_k target[n][k] * Wq[c][k] ; zero the out accumulator
// ---------------------------------------------------------------------------
__global__ void ma_prep(const float* __restrict__ target,
                        const float* __restrict__ Wq,
                        float* __restrict__ vproj,
                        float* __restrict__ outT) {
  int t = blockIdx.x * blockDim.x + threadIdx.x;   // 32768 = N*HD
  int n = t >> 8;
  int c = t & 255;
  float acc = 0.0f;
  #pragma unroll 8
  for (int k = 0; k < D_EMB; ++k)
    acc = fmaf(target[n * D_EMB + k], Wq[c * D_EMB + k], acc);
  vproj[t] = acc;
  outT[t]  = 0.0f;
}

// ---------------------------------------------------------------------------
// fused main kernel: one workgroup = (batch n, 256-row L tile); 4 waves = 4 heads
// ---------------------------------------------------------------------------
__launch_bounds__(128)
__global__ void ma_fused(const float* __restrict__ seq,
                         const float* __restrict__ mem,
                         const float* __restrict__ Wk,
                         const float* __restrict__ Wv,
                         const float* __restrict__ W1,
                         const float* __restrict__ b1,
                         const float* __restrict__ W2,
                         const float* __restrict__ b2,
                         const float* __restrict__ vproj,
                         float* __restrict__ outT,
                         float* __restrict__ attOut) {
  __shared__ unsigned short As[RB * 68];              // bf16 A tile, padded
  __shared__ float Es[H_HEADS][RB][ES_STRIDE];        // f32 E tile (per-wave)
  __shared__ unsigned int BvL[H_HEADS * 2 * 4 * 32 * 8];   // packed Bv frags, 32 KB

  const int tid  = threadIdx.x;
  const int lane = tid & 31;
  const int wave = tid >> 5;        // head index
  const int half = lane >> 4;
  const int ln16 = lane & 15;
  const int n    = blockIdx.y;
  const int l0t  = blockIdx.x * TL;

  // per-lane scorer constants (lane = j for the score fragment columns)
  const float b1lane = (ln16 < 8) ? b1[ln16] : 0.0f;
  const float w2lane = (ln16 < 8) ? W2[ln16] : 0.0f;
  const float b2v    = b2[0];

  // Wk bf16 B fragments register-resident; Wv fragments packed into LDS
  // (written and later read by the same wave -> per-wave DS ordering, no barrier).
  bf16x16 Bk[2][4];
  #pragma unroll
  for (int ks = 0; ks < 2; ++ks) {
    #pragma unroll
    for (int nb = 0; nb < 4; ++nb) {
      union { unsigned int u[8]; bf16x16 v; } tk;
      unsigned int* bvp = &BvL[(((wave * 2) + ks) * 4 + nb) * 256 + lane * 8];
      int col = wave * 64 + nb * 16 + ln16;
      #pragma unroll
      for (int r = 0; r < 8; ++r) {
        int k0 = kmap(r, half) + ks * 32;
        const float2 wk2 = *reinterpret_cast<const float2*>(Wk + col * D_EMB + k0);
        const float2 wv2 = *reinterpret_cast<const float2*>(Wv + col * D_EMB + k0);
        tk.u[r] = pack2_bf16(wk2.x, wk2.y);
        bvp[r]  = pack2_bf16(wv2.x, wv2.y);
      }
      Bk[ks][nb] = tk.v;
    }
  }

  // P^T / Q^T bf16 B fragments (64 x 8, cols 8..15 zero), register-resident.
  // Each lane computes its own coefficients directly (tiny L2-hot arrays):
  //   score_j = sum_d [ msk_d * P[h,j,d] + e_d * Q[h,j,d] ]   (msk*e == e)
  //   P = v*W1[j,d]    + m*W1[j,128+d]
  //   Q = v*W1[j,64+d] + m*W1[j,192+d] - W1[j,d] - W1[j,128+d]
  bf16x16 Pf[2], Qf[2];
  #pragma unroll
  for (int ks = 0; ks < 2; ++ks) {
    union { unsigned int u[8]; bf16x16 v; } tp, tq;
    #pragma unroll
    for (int r = 0; r < 8; ++r) {
      float pv[2] = {0.f, 0.f}, qv[2] = {0.f, 0.f};
      if (ln16 < 8) {
        const int j = ln16;
        #pragma unroll
        for (int t = 0; t < 2; ++t) {
          int d = kmap(r, half) + ks * 32 + t;
          float v   = vproj[n * HD + wave * 64 + d];
          float mm  = mem[n * D_EMB + d];
          float w1a = W1[j * 256 + d];
          float w1b = W1[j * 256 + 64 + d];
          float w1c = W1[j * 256 + 128 + d];
          float w1d = W1[j * 256 + 192 + d];
          pv[t] = fmaf(v, w1a, mm * w1c);
          qv[t] = fmaf(v, w1b, fmaf(mm, w1d, -w1a - w1c));
        }
      }
      tp.u[r] = pack2_bf16(pv[0], pv[1]);
      tq.u[r] = pack2_bf16(qv[0], qv[1]);
    }
    Pf[ks] = tp.v;
    Qf[ks] = tq.v;
  }

  float vacc[4] = {0.f, 0.f, 0.f, 0.f};   // per-lane output column accumulator

  for (int ib = 0; ib < NBLK; ++ib) {
    const int l0 = l0t + ib * RB;
    __syncthreads();   // protect As against previous iteration's readers

    // ---- stage A tile: f32 global -> bf16 LDS (+prefetch next tile) ----
    #pragma unroll
    for (int i = 0; i < 2; ++i) {
      int e4  = tid + i * 128;
      int row = e4 >> 4;
      int c4  = (e4 & 15) * 4;
      const float4 sv = *reinterpret_cast<const float4*>(
          seq + ((size_t)n * L_SEQ + (l0 + row)) * D_EMB + c4);
      unsigned int* dst = reinterpret_cast<unsigned int*>(&As[row * 68 + c4]);
      dst[0] = pack2_bf16(sv.x, sv.y);
      dst[1] = pack2_bf16(sv.z, sv.w);
      if (ib + 1 < NBLK)
        __builtin_prefetch(seq + ((size_t)n * L_SEQ + (l0 + RB + row)) * D_EMB + c4, 0, 3);
    }
    __syncthreads();

    // ---- A fragments (shared across heads) ----
    bf16x16 Af[2];
    #pragma unroll
    for (int ks = 0; ks < 2; ++ks) {
      union { unsigned int u[8]; bf16x16 v; } ta;
      #pragma unroll
      for (int r = 0; r < 8; ++r) {
        int k0 = kmap(r, half) + ks * 32;
        ta.u[r] = *reinterpret_cast<const unsigned int*>(&As[ln16 * 68 + k0]);
      }
      Af[ks] = ta.v;
    }

    // ---- E = A @ Wk_h^T, spill to per-wave LDS tile (transpose medium) ----
    {
      f32x8 Ek[4];
      #pragma unroll
      for (int nb = 0; nb < 4; ++nb) {
        f32x8 c = {0.f,0.f,0.f,0.f,0.f,0.f,0.f,0.f};
        c = __builtin_amdgcn_wmma_f32_16x16x32_bf16(false, Af[0], false, Bk[0][nb], (short)0, c, false, false);
        c = __builtin_amdgcn_wmma_f32_16x16x32_bf16(false, Af[1], false, Bk[1][nb], (short)0, c, false, false);
        Ek[nb] = c;
      }
      #pragma unroll
      for (int nb = 0; nb < 4; ++nb) {
        #pragma unroll
        for (int r = 0; r < 8; ++r) {
          int row = half ? (r + 8) : r;
          Es[wave][row][nb * 16 + ln16] = Ek[nb][r];
        }
      }
    }
    // Es is written and read by the same wave only: per-wave DS ordering
    // (DScnt) suffices, no workgroup barrier needed.

    // ---- scorer GEMM: Sc = Msk @ P^T + E @ Q^T  (bf16 WMMA, N=8 padded) ----
    f32x8 Sc = {0.f,0.f,0.f,0.f,0.f,0.f,0.f,0.f};
    {
      bf16x16 MeA[2], MskA[2];
      #pragma unroll
      for (int ks = 0; ks < 2; ++ks) {
        union { unsigned int u[8]; bf16x16 v; } me, mk;
        #pragma unroll
        for (int r = 0; r < 8; ++r) {
          int k0 = kmap(r, half) + ks * 32;
          const float2 e2 = *reinterpret_cast<const float2*>(&Es[wave][ln16][k0]);
          me.u[r] = pack2_bf16(e2.x, e2.y);                     // msk*e == e
          mk.u[r] = (e2.x != 0.f ? 0x00003F80u : 0u) |          // bf16 1.0 lo
                    (e2.y != 0.f ? 0x3F800000u : 0u);           // bf16 1.0 hi
        }
        MeA[ks]  = me.v;
        MskA[ks] = mk.v;
      }
      Sc = __builtin_amdgcn_wmma_f32_16x16x32_bf16(false, MskA[0], false, Pf[0], (short)0, Sc, false, false);
      Sc = __builtin_amdgcn_wmma_f32_16x16x32_bf16(false, MskA[1], false, Pf[1], (short)0, Sc, false, false);
      Sc = __builtin_amdgcn_wmma_f32_16x16x32_bf16(false, MeA[0],  false, Qf[0], (short)0, Sc, false, false);
      Sc = __builtin_amdgcn_wmma_f32_16x16x32_bf16(false, MeA[1],  false, Qf[1], (short)0, Sc, false, false);
    }

    // ---- V = A @ Wv_h^T (Bv fragments reloaded from LDS, b128 lane-contiguous) ----
    f32x8 Vk[4];
    #pragma unroll
    for (int nb = 0; nb < 4; ++nb) {
      f32x8 c = {0.f,0.f,0.f,0.f,0.f,0.f,0.f,0.f};
      #pragma unroll
      for (int ks = 0; ks < 2; ++ks) {
        union { unsigned int u[8]; bf16x16 v; } tv;
        const unsigned int* bvp = &BvL[(((wave * 2) + ks) * 4 + nb) * 256 + lane * 8];
        #pragma unroll
        for (int r = 0; r < 8; ++r) tv.u[r] = bvp[r];
        c = __builtin_amdgcn_wmma_f32_16x16x32_bf16(false, Af[ks], false, tv.v, (short)0, c, false, false);
      }
      Vk[nb] = c;
    }

    // ---- epilogue: att = sigmoid(W2 . sigmoid(Sc + b1) + b2); accumulate V ----
    #pragma unroll
    for (int r = 0; r < 8; ++r) {
      // Sc[r]: (row = r or r+8 by lane half, j = ln16; cols 8..15 are zero)
      float hm   = sigmoidf_fast(Sc[r] + b1lane);
      float term = hm * w2lane;
      term += __shfl_xor(term, 1, 32);
      term += __shfl_xor(term, 2, 32);
      term += __shfl_xor(term, 4, 32);                 // lanes 0..7 / 16..23 valid
      float att = sigmoidf_fast(term + b2v);
      if (ln16 == 0) {
        int row = half ? (r + 8) : r;
        attOut[((size_t)n * L_SEQ + (l0 + row)) * H_HEADS + wave] = att;
      }
      float attb = __shfl(att, lane & 16, 32);         // broadcast within half
      #pragma unroll
      for (int nb = 0; nb < 4; ++nb)
        vacc[nb] = fmaf(attb, Vk[nb][r], vacc[nb]);
    }
  }

  // combine the two lane-halves (same column, different rows), atomically
  // merge the 4 L-tiles of each batch into the workspace accumulator.
  #pragma unroll
  for (int nb = 0; nb < 4; ++nb) {
    float t = vacc[nb] + __shfl_xor(vacc[nb], 16, 32);
    if (half == 0)
      atomicAdd(&outT[n * HD + wave * 64 + nb * 16 + ln16], t);
  }
}

// ---------------------------------------------------------------------------
// finish: output[n][d] = bu[d] + sum_c outT[n][c] * Wu[d][c]
// ---------------------------------------------------------------------------
__global__ void ma_finish(const float* __restrict__ outT,
                          const float* __restrict__ Wu,
                          const float* __restrict__ bu,
                          float* __restrict__ outp) {
  int t = blockIdx.x * blockDim.x + threadIdx.x;  // 8192 = N*D
  int n = t >> 6;
  int d = t & 63;
  float acc = bu[d];
  #pragma unroll 8
  for (int c = 0; c < HD; ++c)
    acc = fmaf(outT[n * HD + c], Wu[d * HD + c], acc);
  outp[n * D_EMB + d] = acc;
}

extern "C" void kernel_launch(void* const* d_in, const int* in_sizes, int n_in,
                              void* d_out, int out_size, void* d_ws, size_t ws_size,
                              hipStream_t stream) {
  (void)in_sizes; (void)n_in; (void)out_size; (void)ws_size;
  const float* seq = (const float*)d_in[0];   // [128,1024,64]
  const float* tgt = (const float*)d_in[1];   // [128,64]
  const float* mem = (const float*)d_in[2];   // [128,64]
  const float* Wk  = (const float*)d_in[3];   // [256,64]
  const float* Wq  = (const float*)d_in[4];   // [256,64]
  const float* Wv  = (const float*)d_in[5];   // [256,64]
  const float* Wu  = (const float*)d_in[6];   // [64,256]
  const float* bu  = (const float*)d_in[7];   // [64]
  const float* W1  = (const float*)d_in[8];   // [8,256]
  const float* b1  = (const float*)d_in[9];   // [8]
  const float* W2  = (const float*)d_in[10];  // [1,8]
  const float* b2  = (const float*)d_in[11];  // [1]

  float* outp   = (float*)d_out;                       // output [128,64]
  float* attOut = outp + N_BATCH * D_EMB;              // att    [128,1024,4,1]

  float* vproj = (float*)d_ws;                         // [128,256]
  float* outT  = vproj + N_BATCH * HD;                 // [128,256]

  ma_prep<<<dim3(N_BATCH * HD / 256), dim3(256), 0, stream>>>(tgt, Wq, vproj, outT);
  ma_fused<<<dim3(L_SEQ / TL, N_BATCH), dim3(128), 0, stream>>>(
      seq, mem, Wk, Wv, W1, b1, W2, b2, vproj, outT, attOut);
  ma_finish<<<dim3(N_BATCH * D_EMB / 256), dim3(256), 0, stream>>>(outT, Wu, bu, outp);
}
